// CrossAttention_43310450213014
// MI455X (gfx1250) — matrix operand order repaired and checked by reading, not compiled, not run on hardware
//
#include <hip/hip_runtime.h>
#include <hip/hip_bf16.h>
#include <stdint.h>

typedef __attribute__((ext_vector_type(16))) _Float16 v16h;
typedef __attribute__((ext_vector_type(8)))  _Float16 v8h;
typedef __attribute__((ext_vector_type(8)))  float    v8f;
typedef __attribute__((ext_vector_type(4)))  int      v4i;

typedef __attribute__((address_space(1))) v4i gv4i;   // global v4i
typedef __attribute__((address_space(3))) v4i lv4i;   // LDS v4i

#define DEV __device__ __forceinline__

// Problem constants
#define BB   32
#define TFN  2048
#define TPN  256
#define FD   768
#define PD   512
#define AD   512

// LDS row stride (in halves) for staged B tiles: 32 data + 8 pad -> conflict-free
#define BSTRIDE 40
// P-tile row stride in halves
#define PSTRIDE 264

// ---------------- async global->LDS staging ----------------
DEV void stage16(const void* g, void* l) {
#if __has_builtin(__builtin_amdgcn_global_load_async_to_lds_b128)
    __builtin_amdgcn_global_load_async_to_lds_b128(
        (gv4i*)(uintptr_t)g,
        (lv4i*)(uint32_t)(uintptr_t)l,
        0, 0);
#else
    *(v8h*)l = *(const v8h*)g;   // fallback: VGPR bounce (load + ds_store)
#endif
}

template<int N> DEV void wait_async() {
#if __has_builtin(__builtin_amdgcn_s_wait_asynccnt)
    __builtin_amdgcn_s_wait_asynccnt(N);
#else
    asm volatile("s_wait_asynccnt %0" :: "i"(N) : "memory");
#endif
}

// ---------- WMMA fragment builders (CDNA5 16x16x32 f16 layouts) ----------
// A (16x32, MxK): lane sub holds row M=sub; K = k0 + hf*16 + i (contiguous 16 halves).
DEV v16h load_a_f32row(const float* __restrict__ row, int k0, int hf) {
    const float* p = row + k0 + hf * 16;
    v16h a;
#pragma unroll
    for (int i = 0; i < 8; ++i) a[i] = (_Float16)p[i];
#pragma unroll
    for (int i = 0; i < 8; ++i) a[8 + i] = (_Float16)p[8 + i];
    return a;
}

DEV v16h load_a_f16row(const _Float16* row, int k0, int hf) {
    const _Float16* p = row + k0 + hf * 16;
    v8h lo = *(const v8h*)(p);
    v8h hi = *(const v8h*)(p + 8);
    v16h a;
#pragma unroll
    for (int i = 0; i < 8; ++i) { a[i] = lo[i]; a[8 + i] = hi[i]; }
    return a;
}

// B (32x16, KxN): lane sub holds column N=sub; K = k0 + hf*16 + i (contiguous 16 halves).
DEV v16h load_frag16(const _Float16* p) {
    v8h lo = *(const v8h*)(p);
    v8h hi = *(const v8h*)(p + 8);
    v16h b;
#pragma unroll
    for (int i = 0; i < 8; ++i) { b[i] = lo[i]; b[8 + i] = hi[i]; }
    return b;
}

DEV v8f wmma16(v16h a, v16h b, v8f c) {
    return __builtin_amdgcn_wmma_f32_16x16x32_f16(false, a, false, b, (short)0, c, false, false);
}

// ---------------- Kernel 1: weight transpose + f32->f16 ----------------
__global__ void transpose_f16_kernel(const float* __restrict__ W, _Float16* __restrict__ WT,
                                     int R, int C) {
    int idx = blockIdx.x * blockDim.x + threadIdx.x;
    if (idx >= R * C) return;
    int c = idx / R, r = idx % R;                  // WT[c][r] = W[r][c]
    WT[(size_t)c * R + r] = (_Float16)W[(size_t)r * C + c];
}

// ---------------- Projection GEMM: Y = X*W^T + bias ----------------
// 256 threads = 8 waves; each wave owns 16 rows x 512 cols (4 chunks of 8 tiles).
// All waves share the same weight slice -> stage it in LDS, double buffered.
template<int K>
__global__ __launch_bounds__(256) void gemm_proj_kernel(
        const float* __restrict__ X,       // [rows][K] fp32
        const _Float16* __restrict__ WT,   // [512][K] f16
        const float* __restrict__ bias,    // [512]
        float* __restrict__ Yf32,          // [rows][512] or null
        _Float16* __restrict__ Yf16,       // [rows][512] or null
        _Float16* __restrict__ YT16,       // [b][512][rowsPerBatch] or null
        int rowsPerBatch) {
    __shared__ _Float16 wbuf[2][128][BSTRIDE];   // 2 x 10KB

    const int tid = threadIdx.x;
    const int wid = tid >> 5, lane = tid & 31;
    const int sub = lane & 15, hf = lane >> 4;
    const int row0 = (blockIdx.x * 8 + wid) * 16;
    const float* arow = X + (size_t)(row0 + sub) * K;

    constexpr int NST = K / 32;

    auto stage = [&](int buf, int c, int k0) {
        const _Float16* src = WT + (size_t)(c * 128) * K + k0;
        _Float16* dst = &wbuf[buf][0][0];
#pragma unroll
        for (int i = 0; i < 2; ++i) {
            int idx = i * 256 + tid;             // 512 x 16B segments
            int r = idx >> 2, s = idx & 3;
            stage16(src + (size_t)r * K + s * 8, dst + r * BSTRIDE + s * 8);
        }
    };

    stage(0, 0, 0);
    for (int c = 0; c < 4; ++c) {
        v8f acc[8] = {};
        v16h a_cur = load_a_f32row(arow, 0, hf);
        for (int st = 0; st < NST; ++st) {
            int g = c * NST + st;
            if (g < 4 * NST - 1) {
                int gn = g + 1;
                stage(gn & 1, gn / NST, (gn % NST) * 32);
            }
            v16h a_nxt = (st < NST - 1) ? load_a_f32row(arow, (st + 1) * 32, hf) : a_cur;
            if (g < 4 * NST - 1) wait_async<2>(); else wait_async<0>();
            __syncthreads();
            const _Float16* base = &wbuf[g & 1][0][0];
#pragma unroll
            for (int t = 0; t < 8; ++t)
                acc[t] = wmma16(a_cur, load_frag16(base + (t * 16 + sub) * BSTRIDE + hf * 16),
                                acc[t]);
            __syncthreads();
            a_cur = a_nxt;
        }
        // writeback
#pragma unroll
        for (int t = 0; t < 8; ++t) {
            int col = c * 128 + t * 16 + sub;
            float bv = bias[col];
#pragma unroll
            for (int r = 0; r < 8; ++r) {
                int row = row0 + hf * 8 + r;
                float v = acc[t][r] + bv;
                if (Yf32) Yf32[(size_t)row * 512 + col] = v;
                if (Yf16) Yf16[(size_t)row * 512 + col] = (_Float16)v;
                if (YT16) {
                    int b_ = row / rowsPerBatch, p = row % rowsPerBatch;
                    YT16[((size_t)b_ * 512 + col) * rowsPerBatch + p] = (_Float16)v;
                }
            }
        }
    }
}

// ---------------- Fused attention: energy + softmax + P*K ----------------
// 128 threads = 4 waves, all on the same batch; each wave owns 16 frame rows.
// Phase 1 stages K[b] k-slices in LDS (shared 4-way, double buffered, async).
// Phase 2 reuses the LDS arena for the P tiles + staged K^T slices.
__global__ __launch_bounds__(128) void attn_kernel(
        const float* __restrict__ qf32,     // [B*TF][512]
        const _Float16* __restrict__ k16,   // [B][TP][512]
        const _Float16* __restrict__ kT16,  // [B][512][TP]
        const int* __restrict__ mask,       // [B][TP]
        float* __restrict__ out_att,        // [B*TF][1024] (cols 0..511 raw)
        float* __restrict__ out_energy) {   // [B*TF][256]
    __shared__ _Float16 smem[27648];                  // 54KB arena
    _Float16* kbuf  = smem;                           // [2][256][40] (energy phase)
    _Float16* plds  = smem;                           // [4][16][264] (att phase)
    _Float16* ktbuf = smem + 16896;                   // [2][128][40] (att phase)

    const int tid = threadIdx.x;
    const int wid = tid >> 5, lane = tid & 31;
    const int sub = lane & 15, hf = lane >> 4;
    const int b  = blockIdx.x >> 5;                   // 32 blocks per batch
    const int m0 = ((blockIdx.x & 31) * 4 + wid) * 16;

    auto stage_energy = [&](int buf, int k0) {
        const _Float16* src = k16 + (size_t)b * TPN * AD + k0;
        _Float16* dst = kbuf + buf * 256 * BSTRIDE;
#pragma unroll
        for (int i = 0; i < 8; ++i) {
            int idx = i * 128 + tid;                  // 1024 x 16B segments
            int r = idx >> 2, s = idx & 3;
            stage16(src + (size_t)r * AD + s * 8, dst + r * BSTRIDE + s * 8);
        }
    };

    // ---- energy = q @ k^T  (16 x 256 per wave) ----
    const float* qrow = qf32 + (size_t)(b * TFN + m0 + sub) * AD;
    v8f acc[16] = {};
    stage_energy(0, 0);
    v16h a_cur = load_a_f32row(qrow, 0, hf);
    for (int st = 0; st < 16; ++st) {
        if (st < 15) stage_energy((st + 1) & 1, (st + 1) * 32);
        v16h a_nxt = (st < 15) ? load_a_f32row(qrow, (st + 1) * 32, hf) : a_cur;
        if (st < 15) wait_async<8>(); else wait_async<0>();
        __syncthreads();
        const _Float16* base = kbuf + (st & 1) * 256 * BSTRIDE;
#pragma unroll
        for (int t = 0; t < 16; ++t)
            acc[t] = wmma16(a_cur, load_frag16(base + (t * 16 + sub) * BSTRIDE + hf * 16),
                            acc[t]);
        __syncthreads();
        a_cur = a_nxt;
    }

    // kick off K^T staging for phase 2 (overlaps writeback + softmax VALU work)
    auto stage_att = [&](int buf, int c, int kp) {
        const _Float16* src = kT16 + ((size_t)b * AD + c * 128) * TPN + kp;
        _Float16* dst = ktbuf + buf * 128 * BSTRIDE;
#pragma unroll
        for (int i = 0; i < 4; ++i) {
            int idx = i * 128 + tid;                  // 512 x 16B segments
            int r = idx >> 2, s = idx & 3;
            stage16(src + (size_t)r * TPN + s * 8, dst + r * BSTRIDE + s * 8);
        }
    };
    stage_att(0, 0, 0);

    // ---- mask add + write energy output ----
#pragma unroll
    for (int t = 0; t < 16; ++t) {
        int col = t * 16 + sub;
        float mv = (1.0f - (float)mask[b * TPN + col]) * -1000.0f;
#pragma unroll
        for (int r = 0; r < 8; ++r) {
            acc[t][r] += mv;
            int row = m0 + hf * 8 + r;
            out_energy[((size_t)b * TFN + row) * TPN + col] = acc[t][r];
        }
    }

    // ---- softmax over TP=256 (row lives in one 16-lane half) ----
#pragma unroll
    for (int r = 0; r < 8; ++r) {
        float mx = -3.0e38f;
#pragma unroll
        for (int t = 0; t < 16; ++t) mx = fmaxf(mx, acc[t][r]);
#pragma unroll
        for (int off = 1; off < 16; off <<= 1) mx = fmaxf(mx, __shfl_xor(mx, off, 32));
        float s = 0.0f;
#pragma unroll
        for (int t = 0; t < 16; ++t) { float e = __expf(acc[t][r] - mx); acc[t][r] = e; s += e; }
#pragma unroll
        for (int off = 1; off < 16; off <<= 1) s += __shfl_xor(s, off, 32);
        float inv = 1.0f / s;
#pragma unroll
        for (int t = 0; t < 16; ++t)
            plds[(size_t)(wid * 16 + hf * 8 + r) * PSTRIDE + t * 16 + sub] =
                (_Float16)(acc[t][r] * inv);
    }

    // ---- att = P @ k   (16 x 512 per wave, 4 chunks of 128 cols) ----
    for (int c = 0; c < 4; ++c) {
        v8f acc2[8] = {};
        for (int st = 0; st < 8; ++st) {
            int g = c * 8 + st;
            if (g < 31) {
                int gn = g + 1;
                stage_att(gn & 1, gn >> 3, (gn & 7) * 32);
            }
            if (g < 31) wait_async<4>(); else wait_async<0>();
            __syncthreads();
            v16h a = load_a_f16row(plds + (size_t)(wid * 16 + sub) * PSTRIDE, st * 32, hf);
            const _Float16* base = ktbuf + (g & 1) * 128 * BSTRIDE;
#pragma unroll
            for (int t = 0; t < 8; ++t)
                acc2[t] = wmma16(a, load_frag16(base + (t * 16 + sub) * BSTRIDE + hf * 16),
                                 acc2[t]);
            __syncthreads();
        }
#pragma unroll
        for (int t = 0; t < 8; ++t)
#pragma unroll
            for (int r = 0; r < 8; ++r) {
                int row = m0 + hf * 8 + r;
                out_att[((size_t)b * TFN + row) * 1024 + c * 128 + t * 16 + sub] = acc2[t][r];
            }
    }
}

// ---------------- Concat + LayerNorm over 1024 ----------------
__global__ void ln_kernel(const float* __restrict__ qf32,
                          const float* __restrict__ gamma,
                          const float* __restrict__ beta,
                          float* __restrict__ out) {
    int row = blockIdx.x;
    int tid = threadIdx.x;
    const float* att = out + (size_t)row * 1024;
    const float* q   = qf32 + (size_t)row * 512;

    float x[4], s = 0.0f, ss = 0.0f;
#pragma unroll
    for (int i = 0; i < 4; ++i) {
        int j = tid + i * 256;
        float v = (j < 512) ? att[j] : q[j - 512];
        x[i] = v; s += v; ss += v * v;
    }
#pragma unroll
    for (int off = 1; off < 32; off <<= 1) {
        s  += __shfl_xor(s,  off, 32);
        ss += __shfl_xor(ss, off, 32);
    }
    __shared__ float red[2][8];
    int wid = tid >> 5;
    if ((tid & 31) == 0) { red[0][wid] = s; red[1][wid] = ss; }
    __syncthreads();
    float ts = 0.0f, tss = 0.0f;
#pragma unroll
    for (int w = 0; w < 8; ++w) { ts += red[0][w]; tss += red[1][w]; }
    float mu   = ts * (1.0f / 1024.0f);
    float var  = tss * (1.0f / 1024.0f) - mu * mu;
    float rstd = rsqrtf(var + 1e-5f);
#pragma unroll
    for (int i = 0; i < 4; ++i) {
        int j = tid + i * 256;
        out[(size_t)row * 1024 + j] = (x[i] - mu) * rstd * gamma[j] + beta[j];
    }
}

extern "C" void kernel_launch(void* const* d_in, const int* in_sizes, int n_in,
                              void* d_out, int out_size, void* d_ws, size_t ws_size,
                              hipStream_t stream) {
    (void)in_sizes; (void)n_in; (void)out_size; (void)ws_size;
    const float* frame = (const float*)d_in[0];   // [32,2048,768]
    const float* phn   = (const float*)d_in[1];   // [32,256,512]
    const int*   mask  = (const int*)d_in[2];     // [32,256]
    const float* Wq    = (const float*)d_in[3];   // [768,512]
    const float* bq    = (const float*)d_in[4];   // [512]
    const float* Wk    = (const float*)d_in[5];   // [512,512]
    const float* bk    = (const float*)d_in[6];   // [512]
    const float* gamma = (const float*)d_in[7];   // [1024]
    const float* beta  = (const float*)d_in[8];   // [1024]

    char* ws = (char*)d_ws;
    _Float16* WqT  = (_Float16*)(ws);                 // 512*768 f16
    _Float16* WkT  = (_Float16*)(ws + 786432);        // 512*512 f16
    _Float16* k16  = (_Float16*)(ws + 1310720);       // 32*256*512 f16
    _Float16* kT16 = (_Float16*)(ws + 9699328);       // 32*512*256 f16
    float*    qf32 = (float*)   (ws + 18087936);      // 32*2048*512 f32

    float* out_att    = (float*)d_out;                          // [32*2048][1024]
    float* out_energy = out_att + (size_t)BB * TFN * 1024;      // [32*2048][256]

    transpose_f16_kernel<<<(FD * AD + 255) / 256, 256, 0, stream>>>(Wq, WqT, FD, AD);
    transpose_f16_kernel<<<(PD * AD + 255) / 256, 256, 0, stream>>>(Wk, WkT, PD, AD);

    // K projection: 8192 rows / 128 rows-per-block = 64 blocks
    gemm_proj_kernel<PD><<<64, 256, 0, stream>>>(phn, WkT, bk, nullptr, k16, kT16, TPN);
    // Q projection: 65536 rows / 128 rows-per-block = 512 blocks
    gemm_proj_kernel<FD><<<512, 256, 0, stream>>>(frame, WqT, bq, qf32, nullptr, nullptr, TFN);
    // Fused attention: 4096 wave-tiles, 4 waves/block
    attn_kernel<<<1024, 128, 0, stream>>>(qf32, k16, kT16, mask, out_att, out_energy);
    // Concat + LayerNorm: one block per output row
    ln_kernel<<<BB * TFN, 256, 0, stream>>>(qf32, gamma, beta, out_att);
}
